// GenericGraphEncoder_86552180949267
// MI455X (gfx1250) — matrix operand order repaired and checked by reading, not compiled, not run on hardware
//
#include <hip/hip_runtime.h>

// ---- problem dimensions (fixed by the reference) ----
#define NN   10000          // nodes
#define EE   160000         // edges
#define GGR  64             // graphs
#define FIN  128            // input feature dim
#define HD   64             // hidden dim
#define RREL 4              // relations
#define NLAY 13             // 1 + L stacked layers
#define CT   (NLAY * HD)    // 832 concat width

typedef __attribute__((ext_vector_type(16))) __bf16 v16bf;
typedef __attribute__((ext_vector_type(8)))  float  v8f;

// ---- order-preserving float<->uint mapping for atomic segment-max ----
__device__ __forceinline__ unsigned ordf(float x) {
  unsigned u = __float_as_uint(x);
  return (u & 0x80000000u) ? ~u : (u | 0x80000000u);
}
__device__ __forceinline__ float unordf(unsigned u) {
  return __uint_as_float((u & 0x80000000u) ? (u & 0x7FFFFFFFu) : ~u);
}
#define ORD_NEG_INF 0x007FFFFFu   // ordf(-inf)

// =====================================================================
// Kernel 1: per-relation GEMM  xw[r] = A(LDA) x W[r]   via bf16 WMMA.
//
// Block = 128 threads = 4 waves; wave w computes the 16x16 tile at
// rows [16*bx, +16), cols [16*w, +16).  N = 10000 = 625*16 (no tail).
// K is a template parameter (128 or 64) -> K-loop fully unrolls into
// back-to-back v_wmma_f32_16x16x32_bf16.
//
// W[r] (K x 64, fp32) is staged once per block into LDS, transposed and
// converted to bf16:  WT[col][k].  Because the B-fragment order used here is
// K = hh*16 + i for element i, each lane's whole B fragment is one
// contiguous 32-byte LDS read (2 x ds_load_b128).  Row pad of 8 bf16
// (row stride 68 dwords == 4 banks/lane) keeps the reads conflict-free.
// No divergence anywhere -> EXEC all ones for WMMA.
// =====================================================================
template <int K, int LDA>
__global__ void rgat_gemm(const float* __restrict__ A,
                          const float* __restrict__ W,   // [R, K, HD]
                          float* __restrict__ XW)        // [R, NN, HD]
{
  constexpr int KP = K + 8;                 // padded row stride (bf16 units)
  __shared__ __bf16 WT[HD * KP];

  const int r   = blockIdx.y;
  const int tid = threadIdx.x;
  const float* __restrict__ Wr = W + (size_t)r * K * HD;

  // cooperative transpose + f32->bf16 of the weight tile (one-time)
#pragma unroll
  for (int e = tid; e < K * HD; e += 128) { // K*HD % 128 == 0 -> no divergence
    const int k = e >> 6, c = e & 63;       // coalesced global read W[k][c]
    WT[c * KP + k] = (__bf16)Wr[e];
  }
  __syncthreads();

  const int row0 = blockIdx.x * 16;
  const int wave = tid >> 5;                // 0..3 -> column tile
  const int lane = tid & 31;
  const int col0 = wave * 16;
  const int m    = lane & 15;               // row (A) / col (B,D) within tile
  const int hh   = lane >> 4;               // lane half selects K sub-range
  const int col  = col0 + m;

  const float* __restrict__ Ar = A + (size_t)(row0 + m) * LDA;

  v8f acc = {};
#pragma unroll
  for (int k0 = 0; k0 < K; k0 += 32) {
    __builtin_prefetch(Ar + k0 + 32, 0, 0);     // -> global_prefetch_b8
    // A 16x32 bf16 fragment: element i <-> K = hh*16 + i
    v16bf a;
#pragma unroll
    for (int i = 0; i < 8; ++i) {
      a[i]     = (__bf16)Ar[k0 + hh * 16 + i];
      a[i + 8] = (__bf16)Ar[k0 + hh * 16 + 8 + i];
    }
    // B 32x16 fragment: element i <-> K = hh*16 + i  -> contiguous LDS read
    const v16bf b = *(const v16bf*)&WT[col * KP + k0 + hh * 16];
    acc = __builtin_amdgcn_wmma_f32_16x16x32_bf16(
        /*neg_a=*/false, a, /*neg_b=*/false, b,
        /*c_mod=*/(short)0, acc, /*reuse_a=*/false, /*reuse_b=*/false);
  }

  // D 16x16 f32 layout: VGPR i -> M = i + hh*8, N = lane&15
#pragma unroll
  for (int i = 0; i < 8; ++i) {
    const int ro = row0 + i + hh * 8;
    XW[((size_t)r * NN + ro) * HD + col] = acc[i];
  }
}

// =====================================================================
// Kernel 2: per-layer init: segment max/sum state + bias-seeded output
// acc points at node_repr column slice for this layer (stride CT).
// =====================================================================
__global__ void rgat_init(unsigned* __restrict__ mq, float* __restrict__ s,
                          float* __restrict__ acc, const float* __restrict__ bias)
{
  const int idx = blockIdx.x * blockDim.x + threadIdx.x;
  if (idx >= NN * HD) return;
  const int n = idx >> 6, h = idx & 63;
  acc[(size_t)n * CT + h] = bias[h];
  if (h == 0) { mq[n] = ORD_NEG_INF; s[n] = 0.0f; }
}

// =====================================================================
// Kernel 3: edge attention scores + atomic segment max over dst
// =====================================================================
__global__ void rgat_score(const float* __restrict__ XW,
                           const int* __restrict__ src, const int* __restrict__ dst,
                           const int* __restrict__ et,
                           const float* __restrict__ q, const float* __restrict__ k,
                           float* __restrict__ alpha, unsigned* __restrict__ mq)
{
  __shared__ float sq[HD], sk[HD];
  if (threadIdx.x < HD) { sq[threadIdx.x] = q[threadIdx.x]; sk[threadIdx.x] = k[threadIdx.x]; }
  __syncthreads();
  const int e = blockIdx.x * blockDim.x + threadIdx.x;
  if (e >= EE) return;
  const int sN = src[e], dN = dst[e], r = et[e];
  const float4* xi = (const float4*)(XW + ((size_t)r * NN + dN) * HD); // 256B-aligned rows
  const float4* xj = (const float4*)(XW + ((size_t)r * NN + sN) * HD);
  float accv = 0.0f;
#pragma unroll
  for (int h4 = 0; h4 < HD / 4; ++h4) {
    const float4 a4 = xi[h4], b4 = xj[h4];
    const int h = h4 * 4;
    accv += a4.x * sq[h]     + a4.y * sq[h + 1] + a4.z * sq[h + 2] + a4.w * sq[h + 3];
    accv += b4.x * sk[h]     + b4.y * sk[h + 1] + b4.z * sk[h + 2] + b4.w * sk[h + 3];
  }
  const float al = (accv > 0.0f) ? accv : 0.2f * accv;   // leaky_relu, slope 0.2
  alpha[e] = al;
  atomicMax(&mq[dN], ordf(al));
}

// =====================================================================
// Kernel 4: e = exp(alpha - max) ; segment sum over dst
// =====================================================================
__global__ void rgat_expsum(const float* __restrict__ alpha, const int* __restrict__ dst,
                            const unsigned* __restrict__ mq,
                            float* __restrict__ eb, float* __restrict__ s)
{
  const int e = blockIdx.x * blockDim.x + threadIdx.x;
  if (e >= EE) return;
  const int d = dst[e];
  const float mx = unordf(mq[d]);           // finite: d has >= 1 in-edge
  const float ex = __expf(alpha[e] - mx);
  eb[e] = ex;
  atomicAdd(&s[d], ex);
}

// =====================================================================
// Kernel 5: weighted scatter-add of messages, (edge,h)-parallel,
// coalesced over h; writes straight into the node_repr concat slice.
// =====================================================================
__global__ void rgat_aggr(const float* __restrict__ XW,
                          const int* __restrict__ src, const int* __restrict__ dst,
                          const int* __restrict__ et,
                          const float* __restrict__ eb, const float* __restrict__ s,
                          float* __restrict__ acc)
{
  const int idx = blockIdx.x * blockDim.x + threadIdx.x;   // EE*HD = 10.24M exact
  const int e = idx >> 6, h = idx & 63;
  const int d = dst[e];
  const float w = eb[e] / (s[d] + 1e-16f);
  const float v = XW[((size_t)et[e] * NN + src[e]) * HD + h];
  atomicAdd(&acc[(size_t)d * CT + h], w * v);
}

// =====================================================================
// Final softmax-aggregation over batch -> graph_repr [G, CT]
// =====================================================================
__global__ void pool_init(unsigned* __restrict__ mg, float* __restrict__ sg,
                          float* __restrict__ gout)
{
  const int idx = blockIdx.x * blockDim.x + threadIdx.x;
  if (idx >= GGR * CT) return;
  mg[idx] = ORD_NEG_INF; sg[idx] = 0.0f; gout[idx] = 0.0f;
}

__global__ void pool_max(const float* __restrict__ nr, const int* __restrict__ batch,
                         const float* __restrict__ t, unsigned* __restrict__ mg)
{
  const int idx = blockIdx.x * blockDim.x + threadIdx.x;
  if (idx >= NN * CT) return;
  const int n = idx / CT, c = idx % CT;
  atomicMax(&mg[batch[n] * CT + c], ordf(nr[idx] * t[0]));
}

__global__ void pool_sum(const float* __restrict__ nr, const int* __restrict__ batch,
                         const float* __restrict__ t, const unsigned* __restrict__ mg,
                         float* __restrict__ sg)
{
  const int idx = blockIdx.x * blockDim.x + threadIdx.x;
  if (idx >= NN * CT) return;
  const int n = idx / CT, c = idx % CT;
  const int g = batch[n];
  float mx = unordf(mg[g * CT + c]);
  if (mx < -1e38f) mx = 0.0f;               // where(isfinite(m), m, 0)
  atomicAdd(&sg[g * CT + c], __expf(nr[idx] * t[0] - mx));
}

__global__ void pool_wsum(const float* __restrict__ nr, const int* __restrict__ batch,
                          const float* __restrict__ t, const unsigned* __restrict__ mg,
                          const float* __restrict__ sg, float* __restrict__ gout)
{
  const int idx = blockIdx.x * blockDim.x + threadIdx.x;
  if (idx >= NN * CT) return;
  const int n = idx / CT, c = idx % CT;
  const int g = batch[n];
  float mx = unordf(mg[g * CT + c]);
  if (mx < -1e38f) mx = 0.0f;
  const float w = __expf(nr[idx] * t[0] - mx) / (sg[g * CT + c] + 1e-16f);
  atomicAdd(&gout[g * CT + c], w * nr[idx]);
}

// =====================================================================
extern "C" void kernel_launch(void* const* d_in, const int* in_sizes, int n_in,
                              void* d_out, int out_size, void* d_ws, size_t ws_size,
                              hipStream_t stream)
{
  const float* node_features = (const float*)d_in[0];
  const int*   edge_index    = (const int*)  d_in[1];   // [2, E]
  const int*   edge_type     = (const int*)  d_in[2];
  const int*   batch_index   = (const int*)  d_in[3];
  const float* W0  = (const float*)d_in[4];             // [R, FIN, HD]
  const float* q0  = (const float*)d_in[5];
  const float* k0  = (const float*)d_in[6];
  const float* b0  = (const float*)d_in[7];
  const float* Ws  = (const float*)d_in[8];             // [L, R, HD, HD]
  const float* qs  = (const float*)d_in[9];
  const float* kks = (const float*)d_in[10];
  const float* bs  = (const float*)d_in[11];
  const float* t   = (const float*)d_in[12];

  float* graph_repr = (float*)d_out;                    // [G, CT]
  float* node_repr  = (float*)d_out + (size_t)GGR * CT; // [N, CT]

  // workspace carve-up
  char* p = (char*)d_ws;
  float*    XW    = (float*)p;    p += (size_t)RREL * NN * HD * sizeof(float);
  float*    alpha = (float*)p;    p += (size_t)EE * sizeof(float);
  float*    eb    = (float*)p;    p += (size_t)EE * sizeof(float);
  unsigned* mq    = (unsigned*)p; p += (size_t)NN * sizeof(unsigned);
  float*    sN    = (float*)p;    p += (size_t)NN * sizeof(float);
  unsigned* mg    = (unsigned*)p; p += (size_t)GGR * CT * sizeof(unsigned);
  float*    sg    = (float*)p;    p += (size_t)GGR * CT * sizeof(float);

  const int* src = edge_index;
  const int* dst = edge_index + EE;

  for (int l = 0; l < NLAY; ++l) {
    const float *q, *kk, *bias;
    if (l == 0) {
      q = q0; kk = k0; bias = b0;
      rgat_gemm<FIN, FIN><<<dim3(NN / 16, RREL), 128, 0, stream>>>(
          node_features, W0, XW);
    } else {
      q    = qs  + (size_t)(l - 1) * HD;
      kk   = kks + (size_t)(l - 1) * HD;
      bias = bs  + (size_t)(l - 1) * HD;
      rgat_gemm<HD, CT><<<dim3(NN / 16, RREL), 128, 0, stream>>>(
          node_repr + (size_t)(l - 1) * HD,
          Ws + (size_t)(l - 1) * RREL * HD * HD, XW);
    }
    float* acc = node_repr + (size_t)l * HD;   // layer's column slice of concat

    rgat_init  <<<(NN * HD + 255) / 256, 256, 0, stream>>>(mq, sN, acc, bias);
    rgat_score <<<(EE + 255) / 256, 256, 0, stream>>>(XW, src, dst, edge_type, q, kk, alpha, mq);
    rgat_expsum<<<(EE + 255) / 256, 256, 0, stream>>>(alpha, dst, mq, eb, sN);
    rgat_aggr  <<<(EE * HD) / 256, 256, 0, stream>>>(XW, src, dst, edge_type, eb, sN, acc);
  }

  pool_init<<<(GGR * CT + 255) / 256, 256, 0, stream>>>(mg, sg, graph_repr);
  pool_max <<<(NN * CT + 255) / 256, 256, 0, stream>>>(node_repr, batch_index, t, mg);
  pool_sum <<<(NN * CT + 255) / 256, 256, 0, stream>>>(node_repr, batch_index, t, mg, sg);
  pool_wsum<<<(NN * CT + 255) / 256, 256, 0, stream>>>(node_repr, batch_index, t, mg, sg, graph_repr);
}
